// ConditioningNetwork_4776003633741
// MI455X (gfx1250) — matrix operand order repaired and checked by reading, not compiled, not run on hardware
//
#include <hip/hip_runtime.h>

typedef __attribute__((ext_vector_type(16))) _Float16 v16h;
typedef __attribute__((ext_vector_type(8)))  _Float16 v8h;
typedef __attribute__((ext_vector_type(8)))  float    v8f;

union Frag16 {
    v16h v;
    v8h  h[2];
    _Float16 e[16];
};

#define BLOCK 256
#define WAVES 8   // wave32: 256 threads = 8 waves

// exp(-0.5*t*t) == exp2(C*t*t), C = -0.5*log2(e)
#define EXP2C (-0.72134752044448170f)

__global__ __launch_bounds__(BLOCK)
void cond_net_fused(const float* __restrict__ pos,
                    const float* __restrict__ wi,
                    const float* __restrict__ rough,
                    const float* __restrict__ W1, const float* __restrict__ b1,
                    const float* __restrict__ W2, const float* __restrict__ b2,
                    const float* __restrict__ W3, const float* __restrict__ b3,
                    float* __restrict__ out, int nTiles)
{
    // Transposed f16 weights: sW*[n*K + k]
    __shared__ _Float16 sW1[64 * 224];
    __shared__ _Float16 sW2[64 * 64];
    __shared__ _Float16 sW3[64 * 64];
    __shared__ float    sB[192];                 // b1 | b2 | b3
    __shared__ _Float16 sH[WAVES * 16 * 64];     // per-wave staging tile [m][k]

    const int tid = threadIdx.x;

    // ---- one-time weight conversion f32 -> f16, transposed into LDS ----
    for (int i = tid; i < 224 * 64; i += BLOCK) {
        int k = i >> 6, n = i & 63;              // W1 is [224][64] row-major
        sW1[n * 224 + k] = (_Float16)W1[i];
    }
    for (int i = tid; i < 64 * 64; i += BLOCK) {
        int k = i >> 6, n = i & 63;
        sW2[n * 64 + k] = (_Float16)W2[i];
        sW3[n * 64 + k] = (_Float16)W3[i];
    }
    if (tid < 192)
        sB[tid] = (tid < 64) ? b1[tid] : (tid < 128 ? b2[tid - 64] : b3[tid - 128]);
    __syncthreads();

    const int lane = tid & 31;
    const int wave = tid >> 5;
    const int col  = lane & 15;   // N index (B/C/D frags) or M row (A frags)
    const int half = lane >> 4;   // lane-half selects K/M sub-range
    _Float16* hs = sH + wave * 16 * 64;

    const int waveGlobal = blockIdx.x * WAVES + wave;
    const int waveStride = gridDim.x * WAVES;

    for (int tile = waveGlobal; tile < nTiles; tile += waveStride) {
        // ---- load this lane's sample row (lanes L and L+16 share row col) ----
        const int s = tile * 16 + col;
        float xin[7];
        xin[0] = pos[s * 3 + 0]; xin[1] = pos[s * 3 + 1]; xin[2] = pos[s * 3 + 2];
        xin[3] = wi [s * 3 + 0]; xin[4] = wi [s * 3 + 1]; xin[5] = wi [s * 3 + 2];
        xin[6] = rough[s];
        #pragma unroll
        for (int d = 0; d < 7; ++d)
            xin[d] = fminf(fmaxf(xin[d], 0.0f), 1.0f);

        // ================= layer 1: enc[16x224] @ W1[224x64] =================
        v8f acc[4] = {};
        #pragma unroll
        for (int kt = 0; kt < 7; ++kt) {
            // Build A fragment in registers: this lane holds row m=col,
            // K runs: [half*8 .. +7] and [16+half*8 .. +7] within this 32-bin block.
            Frag16 a;
            const float x = xin[kt];
            #pragma unroll
            for (int t = 0; t < 8; ++t) {
                float c0 = (float)(half * 8 + t)      * (1.0f / 31.0f);
                float c1 = (float)(16 + half * 8 + t) * (1.0f / 31.0f);
                float d0 = (x - c0) * 32.0f;
                float d1 = (x - c1) * 32.0f;
                a.e[t]     = (_Float16)__builtin_amdgcn_exp2f(EXP2C * d0 * d0);
                a.e[8 + t] = (_Float16)__builtin_amdgcn_exp2f(EXP2C * d1 * d1);
            }
            #pragma unroll
            for (int nt = 0; nt < 4; ++nt) {
                Frag16 b;  // lane holds column n, K runs [half*8 .. +7] and [16+half*8 .. +7]
                const _Float16* p = sW1 + (nt * 16 + col) * 224 + kt * 32 + half * 8;
                b.h[0] = *(const v8h*)p;
                b.h[1] = *(const v8h*)(p + 16);
                acc[nt] = __builtin_amdgcn_wmma_f32_16x16x32_f16(
                    false, a.v, false, b.v, (short)0, acc[nt], false, false);
            }
        }

        // ---- bias + ReLU + f16, stage to wave-private LDS tile [m][64] ----
        #pragma unroll
        for (int nt = 0; nt < 4; ++nt) {
            float bias = sB[nt * 16 + col];
            #pragma unroll
            for (int r = 0; r < 8; ++r) {
                float v = fmaxf(acc[nt][r] + bias, 0.0f);
                hs[(r + half * 8) * 64 + nt * 16 + col] = (_Float16)v;
            }
        }
        // same-wave DS ops are in-order: no barrier needed (wave-private tile)

        // ================= layer 2: h[16x64] @ W2[64x64] =================
        v8f acc2[4] = {};
        #pragma unroll
        for (int kt = 0; kt < 2; ++kt) {
            Frag16 a;  // A frag: row m=col, runs at kt*32+half*8 and +16
            const _Float16* ap = hs + col * 64 + kt * 32 + half * 8;
            a.h[0] = *(const v8h*)ap;
            a.h[1] = *(const v8h*)(ap + 16);
            #pragma unroll
            for (int nt = 0; nt < 4; ++nt) {
                Frag16 b;
                const _Float16* p = sW2 + (nt * 16 + col) * 64 + kt * 32 + half * 8;
                b.h[0] = *(const v8h*)p;
                b.h[1] = *(const v8h*)(p + 16);
                acc2[nt] = __builtin_amdgcn_wmma_f32_16x16x32_f16(
                    false, a.v, false, b.v, (short)0, acc2[nt], false, false);
            }
        }
        #pragma unroll
        for (int nt = 0; nt < 4; ++nt) {
            float bias = sB[64 + nt * 16 + col];
            #pragma unroll
            for (int r = 0; r < 8; ++r) {
                float v = fmaxf(acc2[nt][r] + bias, 0.0f);
                hs[(r + half * 8) * 64 + nt * 16 + col] = (_Float16)v;
            }
        }

        // ================= layer 3: h[16x64] @ W3[64x64] =================
        v8f acc3[4] = {};
        #pragma unroll
        for (int kt = 0; kt < 2; ++kt) {
            Frag16 a;
            const _Float16* ap = hs + col * 64 + kt * 32 + half * 8;
            a.h[0] = *(const v8h*)ap;
            a.h[1] = *(const v8h*)(ap + 16);
            #pragma unroll
            for (int nt = 0; nt < 4; ++nt) {
                Frag16 b;
                const _Float16* p = sW3 + (nt * 16 + col) * 64 + kt * 32 + half * 8;
                b.h[0] = *(const v8h*)p;
                b.h[1] = *(const v8h*)(p + 16);
                acc3[nt] = __builtin_amdgcn_wmma_f32_16x16x32_f16(
                    false, a.v, false, b.v, (short)0, acc3[nt], false, false);
            }
        }

        // ---- bias + f32 output, coalesced along N ----
        #pragma unroll
        for (int nt = 0; nt < 4; ++nt) {
            float bias = sB[128 + nt * 16 + col];
            #pragma unroll
            for (int r = 0; r < 8; ++r) {
                int m = r + half * 8;
                out[(tile * 16 + m) * 64 + nt * 16 + col] = acc3[nt][r] + bias;
            }
        }
    }
}

extern "C" void kernel_launch(void* const* d_in, const int* in_sizes, int n_in,
                              void* d_out, int out_size, void* d_ws, size_t ws_size,
                              hipStream_t stream) {
    const float* pos   = (const float*)d_in[0];
    const float* wi    = (const float*)d_in[1];
    const float* rough = (const float*)d_in[2];
    const float* W1    = (const float*)d_in[3];
    const float* b1    = (const float*)d_in[4];
    const float* W2    = (const float*)d_in[5];
    const float* b2    = (const float*)d_in[6];
    const float* W3    = (const float*)d_in[7];
    const float* b3    = (const float*)d_in[8];
    // d_in[9] = centers (recomputed in-kernel as b/31)

    const int N = in_sizes[0] / 3;
    const int nTiles = N / 16;

    int grid = 2048;
    if (grid * WAVES > nTiles) grid = (nTiles + WAVES - 1) / WAVES;
    if (grid < 1) grid = 1;

    cond_net_fused<<<grid, BLOCK, 0, stream>>>(pos, wi, rough,
                                               W1, b1, W2, b2, W3, b3,
                                               (float*)d_out, nTiles);
}
